// HashedMLP_83373905150326
// MI455X (gfx1250) — compile-verified, matrix operand order checked by reading, not run on hardware
//
#include <hip/hip_runtime.h>

typedef _Float16 v8h  __attribute__((ext_vector_type(8)));
typedef _Float16 v16h __attribute__((ext_vector_type(16)));
typedef float    v8f  __attribute__((ext_vector_type(8)));

#define N_LEVEL   16
#define N_ENTRIES (1u << 20)
#define HASH_MASK (N_ENTRIES - 1u)
#define HSTR  72   // padded LDS row stride (halves) for activations (K=64 + pad)
#define W0STR 40   // padded LDS row stride for W0^T (K=32 + pad)
#define WSTR  72   // padded LDS row stride for W1^T/W2^T/W3pad (K=64 + pad)

// Assemble a 16-half WMMA fragment register from two contiguous 16B LDS reads.
__device__ __forceinline__ v16h ld_frag(const _Float16* lo, const _Float16* hi) {
    v8h a = *(const v8h*)lo;
    v8h b = *(const v8h*)hi;
    v16h r;
#pragma unroll
    for (int i = 0; i < 8; ++i) { r[i] = a[i]; r[i + 8] = b[i]; }
    return r;
}

// One-shot conversion of fp32 weights -> fp16, transposed to [out][in], into workspace.
__global__ void prep_weights(const float* __restrict__ W0, const float* __restrict__ W1,
                             const float* __restrict__ W2, const float* __restrict__ W3,
                             _Float16* __restrict__ wt) {
    int t = blockIdx.x * blockDim.x + threadIdx.x;
    int stride = gridDim.x * blockDim.x;
    // W0: [32][64] row-major -> W0^T [64][32]
    for (int i = t; i < 64 * 32; i += stride) { int o = i >> 5, k = i & 31; wt[i] = (_Float16)W0[k * 64 + o]; }
    _Float16* w1t = wt + 64 * 32;
    for (int i = t; i < 64 * 64; i += stride) { int o = i >> 6, k = i & 63; w1t[i] = (_Float16)W1[k * 64 + o]; }
    _Float16* w2t = wt + 64 * 32 + 64 * 64;
    for (int i = t; i < 64 * 64; i += stride) { int o = i >> 6, k = i & 63; w2t[i] = (_Float16)W2[k * 64 + o]; }
    _Float16* w3t = wt + 64 * 32 + 2 * 64 * 64;
    for (int i = t; i < 64; i += stride) w3t[i] = (_Float16)W3[i];
}

__global__ __launch_bounds__(256) void hashmlp_kernel(
    const float*  __restrict__ x,        // [B,3]
    const float2* __restrict__ tables,   // [16][2^20] float2
    const _Float16* __restrict__ wt,     // prepped f16 transposed weights
    const float* __restrict__ b0, const float* __restrict__ b1,
    const float* __restrict__ b2, const float* __restrict__ b3,
    float* __restrict__ out)             // [B]
{
    __shared__ __align__(16) _Float16 s_h[256 * HSTR];    // features -> activations, in place
    __shared__ __align__(16) _Float16 s_w0[64 * W0STR];
    __shared__ __align__(16) _Float16 s_w1[64 * WSTR];
    __shared__ __align__(16) _Float16 s_w2[64 * WSTR];
    __shared__ __align__(16) _Float16 s_w3p[16 * WSTR];   // W3 padded to 64x16 B-tile (col 0 live)
    __shared__ float s_b0[64], s_b1[64], s_b2[64];

    const int tid = threadIdx.x;

    // ---- stage f16 weights into padded LDS ----
    {
        const _Float16* w0s = wt;
        const _Float16* w1s = wt + 64 * 32;
        const _Float16* w2s = wt + 64 * 32 + 64 * 64;
        const _Float16* w3s = wt + 64 * 32 + 2 * 64 * 64;
        for (int i = tid; i < 64 * 32; i += 256) s_w0[(i >> 5) * W0STR + (i & 31)] = w0s[i];
        for (int i = tid; i < 64 * 64; i += 256) s_w1[(i >> 6) * WSTR  + (i & 63)] = w1s[i];
        for (int i = tid; i < 64 * 64; i += 256) s_w2[(i >> 6) * WSTR  + (i & 63)] = w2s[i];
        for (int i = tid; i < 16 * 64; i += 256) {
            int n = i >> 6, k = i & 63;
            s_w3p[n * WSTR + k] = (n == 0) ? w3s[k] : (_Float16)0.f;
        }
        if (tid < 64) { s_b0[tid] = b0[tid]; s_b1[tid] = b1[tid]; s_b2[tid] = b2[tid]; }
    }

    // ---- phase A: multi-level hashed-grid encoding (L2-resident gathers) ----
    const int gid = blockIdx.x * 256 + tid;
    const float px = x[3 * gid + 0], py = x[3 * gid + 1], pz = x[3 * gid + 2];
    float res = 16.0f;                 // base grid; ×1.5 per level (exact in fp32)
    _Float16* frow = &s_h[tid * HSTR];
#pragma unroll
    for (int lvl = 0; lvl < N_LEVEL; ++lvl) {
        float gx = px * res, gy = py * res, gz = pz * res;
        float fx = floorf(gx), fy = floorf(gy), fz = floorf(gz);
        float tx = gx - fx, ty = gy - fy, tz = gz - fz;
        unsigned cx = (unsigned)fx, cy = (unsigned)fy, cz = (unsigned)fz;
        const float2* tab = tables + (size_t)lvl * N_ENTRIES;
        float f0 = 0.f, f1 = 0.f;
#pragma unroll
        for (int j = 0; j < 8; ++j) {   // corner j: ox=bit2, oy=bit1, oz=bit0 (matches OFF order)
            unsigned ox = (j >> 2) & 1u, oy = (j >> 1) & 1u, oz = j & 1u;
            unsigned h = (cx + ox) ^ ((cy + oy) * 2654435761u) ^ ((cz + oz) * 805459861u);
            h &= HASH_MASK;
            float2 f = tab[h];
            float w = (ox ? tx : 1.f - tx) * (oy ? ty : 1.f - ty) * (oz ? tz : 1.f - tz);
            f0 += w * f.x; f1 += w * f.y;
        }
        frow[2 * lvl]     = (_Float16)f0;
        frow[2 * lvl + 1] = (_Float16)f1;
        res *= 1.5f;
    }

    __syncthreads();

    // ---- phase B: WMMA MLP; each wave owns rows [32w, 32w+32) = two 16-row tiles ----
    const int lane  = tid & 31;
    const int wave  = tid >> 5;
    const int laneN = lane & 15;
    const int kh    = (lane >> 4) << 3;   // A/B fragment K-subblock: 0 or 8
    const int mBase = (lane >> 4) << 3;   // C fragment: lanes 16-31 hold M=8..15
    const float b3v = b3[0];

#pragma unroll
    for (int tt = 0; tt < 2; ++tt) {
        const int tb = wave * 32 + tt * 16;
        // --- layer 0: [16x32] x [32x64], K=32 == one WMMA per N-tile ---
        {
            const _Float16* ar = &s_h[(tb + laneN) * HSTR];
            v16h A = ld_frag(ar + kh, ar + 16 + kh);
            v8f acc[4];
#pragma unroll
            for (int nt = 0; nt < 4; ++nt) {
                const _Float16* br = &s_w0[(nt * 16 + laneN) * W0STR];
                v16h Bf = ld_frag(br + kh, br + 16 + kh);
                v8f c = {};
                acc[nt] = __builtin_amdgcn_wmma_f32_16x16x32_f16(false, A, false, Bf,
                                                                 (short)0, c, false, false);
            }
#pragma unroll
            for (int nt = 0; nt < 4; ++nt) {
                float bb = s_b0[nt * 16 + laneN];
#pragma unroll
                for (int r = 0; r < 8; ++r) {
                    float v = acc[nt][r] + bb;
                    v = v > 0.f ? v : 0.f;
                    s_h[(tb + mBase + r) * HSTR + nt * 16 + laneN] = (_Float16)v;
                }
            }
        }
        // --- layers 1 & 2: [16x64] x [64x64], K=64 == two chained WMMAs, in place ---
#pragma unroll
        for (int ly = 0; ly < 2; ++ly) {
            const _Float16* w  = ly ? s_w2 : s_w1;
            const float*    bs = ly ? s_b2 : s_b1;
            const _Float16* ar = &s_h[(tb + laneN) * HSTR];
            v16h A0 = ld_frag(ar + kh,      ar + 16 + kh);
            v16h A1 = ld_frag(ar + 32 + kh, ar + 48 + kh);
            v8f acc[4];
#pragma unroll
            for (int nt = 0; nt < 4; ++nt) {
                const _Float16* br = &w[(nt * 16 + laneN) * WSTR];
                v16h B0 = ld_frag(br + kh,      br + 16 + kh);
                v16h B1 = ld_frag(br + 32 + kh, br + 48 + kh);
                v8f c = {};
                c = __builtin_amdgcn_wmma_f32_16x16x32_f16(false, A0, false, B0, (short)0, c, false, false);
                c = __builtin_amdgcn_wmma_f32_16x16x32_f16(false, A1, false, B1, (short)0, c, false, false);
                acc[nt] = c;
            }
#pragma unroll
            for (int nt = 0; nt < 4; ++nt) {
                float bb = bs[nt * 16 + laneN];
#pragma unroll
                for (int r = 0; r < 8; ++r) {
                    float v = acc[nt][r] + bb;
                    v = v > 0.f ? v : 0.f;
                    s_h[(tb + mBase + r) * HSTR + nt * 16 + laneN] = (_Float16)v;
                }
            }
        }
        // --- layer 3: [16x64] x [64x16 padded], column 0 = result ---
        {
            const _Float16* ar = &s_h[(tb + laneN) * HSTR];
            v16h A0 = ld_frag(ar + kh,      ar + 16 + kh);
            v16h A1 = ld_frag(ar + 32 + kh, ar + 48 + kh);
            const _Float16* br = &s_w3p[laneN * WSTR];
            v16h B0 = ld_frag(br + kh,      br + 16 + kh);
            v16h B1 = ld_frag(br + 32 + kh, br + 48 + kh);
            v8f c = {};
            c = __builtin_amdgcn_wmma_f32_16x16x32_f16(false, A0, false, B0, (short)0, c, false, false);
            c = __builtin_amdgcn_wmma_f32_16x16x32_f16(false, A1, false, B1, (short)0, c, false, false);
            // lanes with N==0 hold the output column: lane 0 -> M=0..7, lane 16 -> M=8..15
            if (laneN == 0) {
                float* op = out + blockIdx.x * 256 + tb + mBase;
                float4 lo = make_float4(c[0] + b3v, c[1] + b3v, c[2] + b3v, c[3] + b3v);
                float4 hi = make_float4(c[4] + b3v, c[5] + b3v, c[6] + b3v, c[7] + b3v);
                *(float4*)op       = lo;
                *(float4*)(op + 4) = hi;
            }
        }
    }
}

extern "C" void kernel_launch(void* const* d_in, const int* in_sizes, int n_in,
                              void* d_out, int out_size, void* d_ws, size_t ws_size,
                              hipStream_t stream) {
    const float*  x    = (const float*)d_in[0];
    const float2* tabs = (const float2*)d_in[1];
    const float* W0 = (const float*)d_in[2];
    const float* b0 = (const float*)d_in[3];
    const float* W1 = (const float*)d_in[4];
    const float* b1 = (const float*)d_in[5];
    const float* W2 = (const float*)d_in[6];
    const float* b2 = (const float*)d_in[7];
    const float* W3 = (const float*)d_in[8];
    const float* b3 = (const float*)d_in[9];
    float* out = (float*)d_out;
    _Float16* wt = (_Float16*)d_ws;   // 10304 halves used

    prep_weights<<<16, 256, 0, stream>>>(W0, W1, W2, W3, wt);

    const int B = in_sizes[0] / 3;    // 1,048,576 points
    hashmlp_kernel<<<B / 256, 256, 0, stream>>>(x, tabs, wt, b0, b1, b2, b3, out);
}
